// MCNet_56109452755339
// MI455X (gfx1250) — compile-verified, matrix operand order checked by reading, not run on hardware
//
#include <hip/hip_runtime.h>

// ---------------------------------------------------------------------------
// MCNet plane-sweep + triplet contrast, MI455X (gfx1250, wave32, WMMA + TDM)
// ---------------------------------------------------------------------------
namespace {
constexpr int NLBL = 64;   // NLABEL
constexpr int NG   = 8;    // NGAUSS
constexpr int NS_  = 4;
constexpr int T_   = 2;
constexpr int B_   = 4;
constexpr int C_   = 16;
constexpr int H_   = 48;
constexpr int W_   = 160;
constexpr int MID_ = 32;
constexpr int DT_  = 72;   // NLABEL + NGAUSS
}

typedef __attribute__((ext_vector_type(16))) _Float16     v16h;
typedef __attribute__((ext_vector_type(8)))  float        v8f;
typedef __attribute__((ext_vector_type(4)))  float        fv4;
typedef __attribute__((ext_vector_type(4)))  unsigned int uv4;
typedef __attribute__((ext_vector_type(8)))  unsigned int uv8;

union HU { unsigned int u; _Float16 h[2]; };

__device__ __forceinline__ float bilin(const float* __restrict__ tf,
                                       unsigned int xy, unsigned int wpk) {
    int x0 = (int)(xy & 0xffu);
    int y0 = (int)((xy >> 8) & 0xffu);
    int x1 = x0 + 1; if (x1 > W_ - 1) x1 = W_ - 1;
    int y1 = y0 + 1; if (y1 > H_ - 1) y1 = H_ - 1;
    HU hu; hu.u = wpk;
    float wx = (float)hu.h[0];
    float wy = (float)hu.h[1];
    float f00 = tf[y0 * W_ + x0];
    float f01 = tf[y0 * W_ + x1];
    float f10 = tf[y1 * W_ + x0];
    float f11 = tf[y1 * W_ + x1];
    float top = f00 + (f01 - f00) * wx;
    float bot = f10 + (f11 - f10) * wx;
    return top + (bot - top) * wy;
}

// ---------------------------------------------------------------------------
// Main fused kernel: one block per (n, b, h). 512 threads = 16 waves,
// wave w handles channel c = w. Anchor tile staged via the Tensor Data Mover;
// triplet W-reduction via WMMA x ones-matrix (f32 accumulate).
// ---------------------------------------------------------------------------
__global__ __launch_bounds__(512)
void mcnet_triplet_kernel(const float* __restrict__ ref,
                          const float* __restrict__ target,
                          const float* __restrict__ Kp,
                          const float* __restrict__ iKp,
                          const float* __restrict__ posep,
                          const float* __restrict__ refdp,
                          const float* __restrict__ noisep,
                          float* __restrict__ out_total) {
    const int bx = blockIdx.x;
    const int n  = bx / (B_ * H_);
    const int rb = bx - n * (B_ * H_);
    const int b  = rb / H_;
    const int h  = rb - b * H_;
    const int tid  = threadIdx.x;
    const int lane = tid & 31;
    const int cch  = tid >> 5;          // wave id == channel

    __shared__ __align__(16) float        s_anchor[C_][W_];  // TDM-staged tile
    __shared__ float                      s_q[T_][W_][3];    // projection rays
    __shared__ float                      s_r[T_][3];        // translation
    __shared__ __align__(16) unsigned int s_cxy[T_][16][W_]; // packed x0|y0<<8
    __shared__ __align__(16) unsigned int s_cw [T_][16][W_]; // packed f16 wx,wy
    __shared__ float                      s_S[C_][80];       // sum sq diff per d

    const float* Kb  = Kp  + b * 16;
    const float* iKb = iKp + b * 16;

    // --- per-block ray precompute: q_j(w) = (K@pose_j)[:3,:3] @ (invK3 @ [w,h,1])
    for (int t = tid; t < T_ * W_; t += blockDim.x) {
        const int j = t / W_;
        const int w = t - j * W_;
        const float* Tb = posep + (j * B_ + b) * 16;
        float P[3][4];
        #pragma unroll
        for (int i = 0; i < 3; ++i)
            #pragma unroll
            for (int k = 0; k < 4; ++k) {
                float s = 0.f;
                #pragma unroll
                for (int l = 0; l < 4; ++l) s += Kb[i * 4 + l] * Tb[l * 4 + k];
                P[i][k] = s;
            }
        const float vx = iKb[0] * (float)w + iKb[1] * (float)h + iKb[2];
        const float vy = iKb[4] * (float)w + iKb[5] * (float)h + iKb[6];
        const float vz = iKb[8] * (float)w + iKb[9] * (float)h + iKb[10];
        #pragma unroll
        for (int i = 0; i < 3; ++i)
            s_q[j][w][i] = P[i][0] * vx + P[i][1] * vy + P[i][2] * vz;
        if (w == 0) { s_r[j][0] = P[0][3]; s_r[j][1] = P[1][3]; s_r[j][2] = P[2][3]; }
    }

    // --- TDM: DMA the 16x160 f32 anchor tile (row stride H*W) into LDS -----
    // D# group0: count=1, lds_addr, global_addr, type=2 ("image")
    // D# group1: data_size=4B, tensor_dim0=W, tensor_dim1=C, tile=WxC,
    //            tensor_dim0_stride=H*W. 2D tensor => 2-operand form (VADDR2/3 NULL)
    if (tid < 32) {
        const float* gsrc = ref + ((((size_t)n * B_ + b) * C_ + 0) * H_ + h) * (size_t)W_;
        const unsigned long long ga = (unsigned long long)(const void*)gsrc;
        const unsigned int lds_addr =
            (unsigned int)(unsigned long long)(&s_anchor[0][0]);
        uv4 g0;
        g0[0] = 1u;                                            // valid, user D#
        g0[1] = lds_addr;                                      // LDS byte addr
        g0[2] = (unsigned int)(ga & 0xFFFFFFFFu);              // global_addr lo
        g0[3] = (unsigned int)((ga >> 32) & 0x1FFFFFFu) | (2u << 30); // hi|type
        uv8 g1;
        g1[0] = 2u << 16;                                      // data_size=4B
        g1[1] = ((unsigned int)W_) << 16;                      // tensor_dim0 lo16
        g1[2] = ((unsigned int)C_) << 16;                      // dim0 hi | dim1 lo
        g1[3] = ((unsigned int)W_) << 16;                      // dim1 hi | tile0
        g1[4] = (unsigned int)C_;                              // tile1 | tile2=0
        g1[5] = (unsigned int)(H_ * W_);                       // dim0_stride lo32
        g1[6] = 0u;                                            // stride hi | d1s
        g1[7] = 0u;
        asm volatile("tensor_load_to_lds %0, %1" :: "s"(g0), "s"(g1) : "memory");
        __builtin_amdgcn_s_wait_tensorcnt(0);
    }

    const float* tf0 = target + ((((size_t)(n * T_ + 0) * B_ + b) * C_ + cch) * H_) * (size_t)W_;
    const float* tf1 = target + ((((size_t)(n * T_ + 1) * B_ + b) * C_ + cch) * H_) * (size_t)W_;
    __builtin_prefetch(tf0 + (size_t)h * W_, 0, 3);
    __builtin_prefetch(tf1 + (size_t)h * W_, 0, 3);

    // B operand: all-ones 32x16 f16 matrix (layout-independent row-sum trick)
    v16h bones;
    #pragma unroll
    for (int i = 0; i < 16; ++i) bones[i] = (_Float16)1.0f;

    const int  m  = lane & 15;          // row within d-tile
    const bool hi = lane >= 16;

    for (int dtile = 0; dtile < 5; ++dtile) {
        __syncthreads();   // previous tile's coeff reads complete / TDM visible

        // --- cooperative bilinear-coefficient precompute for this d-tile ---
        for (int e = tid; e < T_ * 16 * W_; e += blockDim.x) {
            const int j   = e / (16 * W_);
            const int rem = e - j * 16 * W_;
            const int mm  = rem / W_;
            const int w   = rem - mm * W_;
            const int d   = dtile * 16 + mm;
            const float q0 = s_q[j][w][0], q1 = s_q[j][w][1], q2 = s_q[j][w][2];
            const float r0 = s_r[j][0],    r1 = s_r[j][1],    r2 = s_r[j][2];
            const float rd = refdp[((size_t)b * H_ + h) * W_ + w];
            float px, py;
            if (d < NLBL) {
                const float depth = fminf(fmaxf(rd + (float)(d - MID_), 0.f), 255.f);
                const float iz = 1.0f / (depth * q2 + r2 + 1e-7f);
                px = (depth * q0 + r0) * iz;
                py = (depth * q1 + r1) * iz;
            } else {
                const int g = d - NLBL;
                const float depth = fminf(fmaxf(rd, 0.f), 255.f);
                const float iz = 1.0f / (depth * q2 + r2 + 1e-7f);
                px = (depth * q0 + r0) * iz;
                py = (depth * q1 + r1) * iz;
                if (g < NG) {
                    const float* np = noisep +
                        ((((((size_t)n * T_ + j) * NG + g) * B_ + b) * H_ + h) * W_ + w) * 2;
                    px += np[0] * 0.5f * (float)(W_ - 1);
                    py += np[1] * 0.5f * (float)(H_ - 1);
                }
            }
            px = fminf(fmaxf(px, 0.f), (float)(W_ - 1));
            py = fminf(fmaxf(py, 0.f), (float)(H_ - 1));
            const float fx0 = floorf(px), fy0 = floorf(py);
            const int x0 = (int)fx0, y0 = (int)fy0;
            HU hu; hu.h[0] = (_Float16)(px - fx0); hu.h[1] = (_Float16)(py - fy0);
            s_cxy[j][mm][w] = (unsigned int)(x0 | (y0 << 8));
            s_cw [j][mm][w] = hu.u;
        }
        __syncthreads();

        // --- per-wave WMMA reduction over W: S[m] = sum_w (a + eps - v)^2 ---
        const int   d    = dtile * 16 + m;
        const float mask = (d < DT_) ? 1.0f : 0.0f;   // branchless label padding
        v8f acc = {};
        #pragma unroll
        for (int kb = 0; kb < W_ / 32; ++kb) {
            const int wbase = kb * 32 + (hi ? 16 : 0);
            const uv4* pxy0 = (const uv4*)&s_cxy[0][m][wbase];
            const uv4* pcw0 = (const uv4*)&s_cw [0][m][wbase];
            const uv4* pxy1 = (const uv4*)&s_cxy[1][m][wbase];
            const uv4* pcw1 = (const uv4*)&s_cw [1][m][wbase];
            const fv4* pan  = (const fv4*)&s_anchor[cch][wbase];
            v16h A;
            #pragma unroll
            for (int q = 0; q < 4; ++q) {             // 128-bit DS loads
                const uv4 xy0 = pxy0[q], cw0 = pcw0[q];
                const uv4 xy1 = pxy1[q], cw1 = pcw1[q];
                const fv4 an  = pan[q];
                #pragma unroll
                for (int r = 0; r < 4; ++r) {
                    const float v0 = bilin(tf0, xy0[r], cw0[r]);
                    const float v1 = bilin(tf1, xy1[r], cw1[r]);
                    const float diff = an[r] + 1e-6f - 0.5f * (v0 + v1);
                    A[q * 4 + r] = (_Float16)(diff * diff * mask);
                }
            }
            // D = A(16x32 f16) x ones(32x16) + C  => row sums, f32 accumulate
            acc = __builtin_amdgcn_wmma_f32_16x16x32_f16(
                false, A, false, bones, (short)0, acc, false, false);
        }
        // D layout: lane 0 column holds M=0..7 in vgpr 0..7; lane 16 holds M=8..15
        if (lane == 0) {
            #pragma unroll
            for (int r = 0; r < 8; ++r) s_S[cch][dtile * 16 + r] = acc[r];
        } else if (lane == 16) {
            #pragma unroll
            for (int r = 0; r < 8; ++r) s_S[cch][dtile * 16 + 8 + r] = acc[r];
        }
    }
    __syncthreads();

    // --- triplet margin loss over the 72 labels (lane-parallel) ---
    const float dap = sqrtf(s_S[cch][MID_]);
    float part = 0.f;
    for (int d = lane; d < DT_; d += 32)
        if (d != MID_)
            part += fmaxf(dap - sqrtf(s_S[cch][d]) + 1.0f, 0.0f);
    #pragma unroll
    for (int off = 16; off > 0; off >>= 1)
        part += __shfl_xor(part, off, 32);
    if (lane == 0)
        out_total[(((size_t)n * B_ + b) * C_ + cch) * H_ + h] = part / (float)(DT_ - 1);
}

// ---------------------------------------------------------------------------
// cost_sum[MID] for n = NS-1 (depth offset 0): plain VALU kernel
// ---------------------------------------------------------------------------
__global__ __launch_bounds__(256)
void mcnet_cost_kernel(const float* __restrict__ ref,
                       const float* __restrict__ target,
                       const float* __restrict__ Kp,
                       const float* __restrict__ iKp,
                       const float* __restrict__ posep,
                       const float* __restrict__ refdp,
                       float* __restrict__ out_cost) {
    const int idx = blockIdx.x * blockDim.x + threadIdx.x;
    if (idx >= B_ * H_ * W_) return;
    const int w = idx % W_;
    const int h = (idx / W_) % H_;
    const int b = idx / (W_ * H_);
    const int n = NS_ - 1;

    const float* Kb  = Kp  + b * 16;
    const float* iKb = iKp + b * 16;
    const float vx = iKb[0] * (float)w + iKb[1] * (float)h + iKb[2];
    const float vy = iKb[4] * (float)w + iKb[5] * (float)h + iKb[6];
    const float vz = iKb[8] * (float)w + iKb[9] * (float)h + iKb[10];
    const float depth = fminf(fmaxf(refdp[idx], 0.f), 255.f);

    float acc = 0.f;
    for (int j = 0; j < T_; ++j) {
        const float* Tb = posep + (j * B_ + b) * 16;
        float P[3][4];
        #pragma unroll
        for (int i = 0; i < 3; ++i)
            #pragma unroll
            for (int k = 0; k < 4; ++k) {
                float s = 0.f;
                #pragma unroll
                for (int l = 0; l < 4; ++l) s += Kb[i * 4 + l] * Tb[l * 4 + k];
                P[i][k] = s;
            }
        const float q0 = P[0][0] * vx + P[0][1] * vy + P[0][2] * vz;
        const float q1 = P[1][0] * vx + P[1][1] * vy + P[1][2] * vz;
        const float q2 = P[2][0] * vx + P[2][1] * vy + P[2][2] * vz;
        const float iz = 1.0f / (depth * q2 + P[2][3] + 1e-7f);
        float px = fminf(fmaxf((depth * q0 + P[0][3]) * iz, 0.f), (float)(W_ - 1));
        float py = fminf(fmaxf((depth * q1 + P[1][3]) * iz, 0.f), (float)(H_ - 1));
        const float fx0 = floorf(px), fy0 = floorf(py);
        const int x0 = (int)fx0, y0 = (int)fy0;
        int x1 = x0 + 1; if (x1 > W_ - 1) x1 = W_ - 1;
        int y1 = y0 + 1; if (y1 > H_ - 1) y1 = H_ - 1;
        const float wx = px - fx0, wy = py - fy0;
        for (int c = 0; c < C_; ++c) {
            const float* tf = target +
                ((((size_t)(n * T_ + j) * B_ + b) * C_ + c) * H_) * (size_t)W_;
            const float f00 = tf[y0 * W_ + x0];
            const float f01 = tf[y0 * W_ + x1];
            const float f10 = tf[y1 * W_ + x0];
            const float f11 = tf[y1 * W_ + x1];
            const float top = f00 + (f01 - f00) * wx;
            const float bot = f10 + (f11 - f10) * wx;
            const float s   = top + (bot - top) * wy;
            const float rv  = ref[((((size_t)n * B_ + b) * C_ + c) * H_ + h) * W_ + w];
            acc += fabsf(rv - s);
        }
    }
    out_cost[idx] = acc / (float)(C_ * T_);
}

// ---------------------------------------------------------------------------
extern "C" void kernel_launch(void* const* d_in, const int* in_sizes, int n_in,
                              void* d_out, int out_size, void* d_ws, size_t ws_size,
                              hipStream_t stream) {
    (void)in_sizes; (void)n_in; (void)out_size; (void)d_ws; (void)ws_size;
    const float* ref    = (const float*)d_in[0];   // (NS,B,C,H,W)
    const float* target = (const float*)d_in[1];   // (NS,T,B,C,H,W)
    const float* realf  = (const float*)d_in[2];   // (B,C,H,W)
    const float* Kp     = (const float*)d_in[3];   // (B,4,4)
    const float* iKp    = (const float*)d_in[4];   // (B,4,4)
    const float* posep  = (const float*)d_in[5];   // (T,B,4,4)
    const float* refdp  = (const float*)d_in[6];   // (B,1,H,W)
    const float* noisep = (const float*)d_in[7];   // (NS,T,NG,B,H,W,2)

    float* out = (float*)d_out;
    const size_t FEA = (size_t)B_ * C_ * H_ * W_;          // 491520
    float* out_total = out;                                // (NS,B,C,H) = 12288
    float* out_fea0  = out_total + (size_t)NS_ * B_ * C_ * H_;
    float* out_real  = out_fea0 + FEA;
    float* out_cost  = out_real + FEA;                     // (B,H,W) = 30720
    float* out_refN  = out_cost + (size_t)B_ * H_ * W_;

    // passthrough outputs (device-to-device, graph-capture safe)
    hipMemcpyAsync(out_fea0, ref,  FEA * sizeof(float), hipMemcpyDeviceToDevice, stream);
    hipMemcpyAsync(out_real, realf, FEA * sizeof(float), hipMemcpyDeviceToDevice, stream);
    hipMemcpyAsync(out_refN, ref + (size_t)(NS_ - 1) * FEA,
                   FEA * sizeof(float), hipMemcpyDeviceToDevice, stream);

    dim3 g1(NS_ * B_ * H_), b1(512);
    mcnet_triplet_kernel<<<g1, b1, 0, stream>>>(ref, target, Kp, iKp, posep,
                                                refdp, noisep, out_total);
    const int np = B_ * H_ * W_;
    mcnet_cost_kernel<<<(np + 255) / 256, 256, 0, stream>>>(ref, target, Kp, iKp,
                                                            posep, refdp, out_cost);
}
